// Decoder_44787918962879
// MI455X (gfx1250) — matrix operand order repaired and checked by reading, not compiled, not run on hardware
//
#include <hip/hip_runtime.h>

// ---------------------------------------------------------------------------
// Types / helpers
// ---------------------------------------------------------------------------
typedef unsigned short u16;
typedef unsigned int   u32;
typedef __attribute__((ext_vector_type(16))) __bf16 v16bf;
typedef __attribute__((ext_vector_type(8)))  float  v8f;
typedef __attribute__((ext_vector_type(8)))  u32    v8u;

#define DEV __device__ __forceinline__

DEV u16 f2bf(float f) {                       // f32 -> bf16 RNE
  u32 u = __builtin_bit_cast(u32, f);
  u32 r = u + 0x7FFFu + ((u >> 16) & 1u);
  return (u16)(r >> 16);
}
DEV float bf2f(u16 h) { return __builtin_bit_cast(float, ((u32)h) << 16); }
DEV float sigm(float x) { return 1.0f / (1.0f + __expf(-x)); }

// A-matrix (16x32 bf16, M x K) per ISA 7.12.2: lane holds row m=lane&15,
// half=lane>>4 selects K sub-block; VGPR v<4: K=half*8+2v(,+1); v>=4: +16.
DEV v16bf ld_a(const u16* row, int kb, int half) {
  v8u u;
#pragma unroll
  for (int v = 0; v < 4; ++v) {
    int k0 = kb + half * 8 + 2 * v;
    u[v]     = *(const u32*)(row + k0);
    u[v + 4] = *(const u32*)(row + k0 + 16);
  }
  return __builtin_bit_cast(v16bf, u);
}
// B-matrix (32x16 bf16, K x N), weights stored row-major [N][K]:
// lane holds col n=lane&15, kh=lane>>4: VGPR v<4 holds K = kh*8 + 2v (,+1); v>=4: +16.
DEV v16bf ld_b(const u16* row, int kb, int kh) {
  v8u u;
#pragma unroll
  for (int v = 0; v < 8; ++v)
    u[v] = *(const u32*)(row + kb + kh * 8 + 2 * (v & 3) + (v >> 2) * 16);
  return __builtin_bit_cast(v16bf, u);
}
DEV v8f wmma_bf16(v16bf a, v16bf b, v8f c) {
  return __builtin_amdgcn_wmma_f32_16x16x32_bf16(false, a, false, b,
                                                 (short)0, c, false, false);
}

// ---------------------------------------------------------------------------
// Generic WMMA GEMM: D[M,N] = A[M,K](bf16) * B[N,K]^T(bf16) + bias, epilogues
// EPI 0: f32 store | 1: bf16 store | 2: relu->bf16 | 3: relu->bf16 p and q=p+spk
// EPI 4: mel output transpose-scatter into d_out[B,T,80]
// ---------------------------------------------------------------------------
template <int EPI>
__global__ void __launch_bounds__(256)
gemm_bf16_kernel(const u16* __restrict__ A, int lda,
                 const u16* __restrict__ Bw, int ldb,
                 const float* __restrict__ bias,
                 float* __restrict__ outF, u16* __restrict__ out1,
                 u16* __restrict__ out2, const float* __restrict__ aux,
                 int M, int N, int K, int ldo) {
  const int gw   = (blockIdx.x * 256 + threadIdx.x) >> 5;
  const int lane = threadIdx.x & 31;
  const int tilesN = N >> 4;
  const int total  = (M >> 4) * tilesN;
  if (gw >= total) return;                       // wave-uniform exit
  const int mt = gw / tilesN, nt = gw - mt * tilesN;
  const int half = lane >> 4, li = lane & 15;
  const u16* arow = A  + (size_t)(mt * 16 + li) * lda;
  const u16* brow = Bw + (size_t)(nt * 16 + li) * ldb;
  v8f acc = {0.f, 0.f, 0.f, 0.f, 0.f, 0.f, 0.f, 0.f};
  for (int kb = 0; kb < K; kb += 32)
    acc = wmma_bf16(ld_a(arow, kb, half), ld_b(brow, kb, half), acc);
  const int n = nt * 16 + li;
  const float bs = bias ? bias[n] : 0.0f;
#pragma unroll
  for (int r = 0; r < 8; ++r) {
    const int m = mt * 16 + r + 8 * half;        // C layout: VGPR r, half
    float v = acc[r] + bs;
    if (EPI == 0) {
      outF[(size_t)m * ldo + n] = v;
    } else if (EPI == 1) {
      out1[(size_t)m * ldo + n] = f2bf(v);
    } else if (EPI == 2) {
      out1[(size_t)m * ldo + n] = f2bf(fmaxf(v, 0.f));
    } else if (EPI == 3) {
      float rv = fmaxf(v, 0.f);
      out1[(size_t)m * ldo + n] = f2bf(rv);
      out2[(size_t)m * ldo + n] = f2bf(rv + aux[(m & 15) * 1024 + n]);
    } else if (EPI == 4) {                        // row m = t*16+b
      int t = m >> 4, b = m & 15;
      outF[(size_t)(b * 256 + t) * 80 + n] = v;
    }
  }
}

// ---------------------------------------------------------------------------
// Small preprocessing kernels (grid-stride except spk/stop, deterministic)
// ---------------------------------------------------------------------------
__global__ void cvt_kernel(const float* __restrict__ s, u16* __restrict__ d, int n) {
  for (int i = blockIdx.x * 256 + threadIdx.x; i < n; i += gridDim.x * 256)
    d[i] = f2bf(s[i]);
}
__global__ void split_ih1_kernel(const float* __restrict__ src,
                                 u16* __restrict__ dp, u16* __restrict__ dc, int n) {
  for (int i = blockIdx.x * 256 + threadIdx.x; i < n; i += gridDim.x * 256) {
    int row = i / 1280, col = i - row * 1280;
    u16 v = f2bf(src[i]);
    if (col < 1024) dp[row * 1024 + col] = v;
    else            dc[row * 256 + (col - 1024)] = v;
  }
}
__global__ void pad_pre1_kernel(const float* __restrict__ W, u16* __restrict__ d) {
  for (int i = blockIdx.x * 256 + threadIdx.x; i < 1024 * 96; i += gridDim.x * 256) {
    int r = i / 96, c = i - r * 96;
    d[i] = (c < 80) ? f2bf(W[r * 80 + c]) : (u16)0;
  }
}
__global__ void prevmel_kernel(const float* __restrict__ mels, u16* __restrict__ d) {
  for (int i = blockIdx.x * 256 + threadIdx.x; i < 4096 * 96; i += gridDim.x * 256) {
    int row = i / 96, c = i - row * 96;
    int t = row >> 4, b = row & 15;
    float v = (t > 0 && c < 80) ? mels[(b * 256 + t) * 80 + c] : 0.f;
    d[i] = f2bf(v);
  }
}
__global__ void bias_sum_kernel(const float* a, const float* b, float* o, int n) {
  for (int i = blockIdx.x * 256 + threadIdx.x; i < n; i += gridDim.x * 256)
    o[i] = a[i] + b[i];
}
__global__ void spk_kernel(const float* __restrict__ Wsp, const float* __restrict__ bsp,
                           const float* __restrict__ emb, float* __restrict__ spk) {
  int i = blockIdx.x * 256 + threadIdx.x;
  if (i >= 16 * 1024) return;
  int b = i >> 10, n = i & 1023;
  float s = bsp[n];
  for (int k = 0; k < 256; ++k) s += Wsp[n * 256 + k] * emb[b * 256 + k];
  spk[i] = s;
}
__global__ void zero_kernel(int* p, int n) {
  for (int i = blockIdx.x * 256 + threadIdx.x; i < n; i += gridDim.x * 256) p[i] = 0;
}
__global__ void stop_kernel(const u16* __restrict__ H2, const float* __restrict__ Wstop,
                            const float* __restrict__ bstop, float* __restrict__ out) {
  int i = blockIdx.x * 256 + threadIdx.x;    // i = t*16+b
  if (i >= 4096) return;
  int t = i >> 4, b = i & 15;
  float s = bstop[0];
  for (int k = 0; k < 1024; ++k) s += bf2f(H2[(size_t)i * 1024 + k]) * Wstop[k];
  out[16 * 256 * 80 + b * 256 + t] = sigm(s);
}

// ---------------------------------------------------------------------------
// Persistent sequential decoder: 64 blocks x 256 threads, grid barriers
// ---------------------------------------------------------------------------
DEV void gsync(int* bar) {
  __syncthreads();
  if (threadIdx.x == 0) {
    __threadfence();
    volatile int* vgen = bar + 1;
    int g = *vgen;
    if (atomicAdd(bar, 1) == 63) {             // last of 64 blocks
      *bar = 0;
      __threadfence();
      atomicAdd(bar + 1, 1);
    } else {
      while (*vgen == g) __builtin_amdgcn_s_sleep(1);
    }
    __threadfence();
  }
  __syncthreads();
}

struct DecArgs {
  const float *enc, *Wc, *Wloc, *vatt, *qW, *procmem, *bias2;
  const u16 *G1p, *Wih1c, *Whh1, *Wih2, *Whh2;
  float *c1, *c2, *prevw, *cumw, *g1, *g2;
  u16 *h1b, *h2b, *ctxb, *H2;
  int *bar;
};

__global__ void __launch_bounds__(256) decoder_kernel(DecArgs Aa) {
  __shared__ float s_pw[256], s_cw[256], s_red[256], s_w[256];
  const int blk = blockIdx.x, tid = threadIdx.x;
  const int lane = tid & 31;
  const int gw = blk * 8 + (tid >> 5);
  const int half = lane >> 4, li = lane & 15;

  for (int t = 0; t < 256; ++t) {
    // ---- Phase A: location-sensitive attention (blocks 0..15; b=blk, l=tid)
    if (blk < 16) {
      const int b = blk, l = tid;
      s_pw[l] = Aa.prevw[b * 256 + l];
      s_cw[l] = Aa.cumw[b * 256 + l];
      __syncthreads();
      float fl[32];
      for (int c = 0; c < 32; ++c) {
        float s = 0.f;
        const float* w0 = Aa.Wc + c * 62;
        for (int k = 0; k < 31; ++k) {
          int ll = l + k - 15;
          if (ll >= 0 && ll < 256) s += w0[k] * s_pw[ll] + w0[31 + k] * s_cw[ll];
        }
        fl[c] = s;
      }
      const float* qr = Aa.qW + (t * 16 + b) * 128;
      const float* pr = Aa.procmem + (b * 256 + l) * 128;
      float e = 0.f;
      for (int a = 0; a < 128; ++a) {
        float x = qr[a] + pr[a];
        const float* wl = Aa.Wloc + a * 32;
#pragma unroll
        for (int c = 0; c < 32; ++c) x += wl[c] * fl[c];
        e += Aa.vatt[a] * tanhf(x);
      }
      s_red[l] = e; __syncthreads();
      for (int off = 128; off; off >>= 1) {
        if (l < off) s_red[l] = fmaxf(s_red[l], s_red[l + off]);
        __syncthreads();
      }
      const float mx = s_red[0]; __syncthreads();
      const float ex = __expf(e - mx);
      s_red[l] = ex; __syncthreads();
      for (int off = 128; off; off >>= 1) {
        if (l < off) s_red[l] += s_red[l + off];
        __syncthreads();
      }
      const float w = ex / s_red[0];
      s_w[l] = w;
      Aa.prevw[b * 256 + l] = w;
      Aa.cumw[b * 256 + l]  = s_cw[l] + w;
      __syncthreads();
      float cacc = 0.f;                               // ctx[b][d], d=l
      for (int ll = 0; ll < 256; ++ll)
        cacc += s_w[ll] * Aa.enc[(b * 256 + ll) * 256 + l];
      Aa.ctxb[b * 256 + l] = f2bf(cacc);
    }
    gsync(Aa.bar);

    // ---- Phase B1: LSTM1 gates = G1p + ctx*Wih1c^T + h1*Whh1^T  (WMMA)
    if (gw < 256) {
      const int n = gw * 16 + li;
      v8f acc;
#pragma unroll
      for (int r = 0; r < 8; ++r)
        acc[r] = bf2f(Aa.G1p[(size_t)(t * 16 + r + 8 * half) * 4096 + n]);
      const u16* ar = Aa.ctxb + li * 256;
      const u16* br = Aa.Wih1c + (size_t)n * 256;
      for (int kb = 0; kb < 256; kb += 32)
        acc = wmma_bf16(ld_a(ar, kb, half), ld_b(br, kb, half), acc);
      const u16* ar2 = Aa.h1b + li * 1024;
      const u16* br2 = Aa.Whh1 + (size_t)n * 1024;
      __builtin_prefetch(br2, 0, 1);
      for (int kb = 0; kb < 1024; kb += 32)
        acc = wmma_bf16(ld_a(ar2, kb, half), ld_b(br2, kb, half), acc);
#pragma unroll
      for (int r = 0; r < 8; ++r) Aa.g1[(r + 8 * half) * 4096 + n] = acc[r];
    }
    gsync(Aa.bar);

    // ---- Phase B2: LSTM1 pointwise (16384 threads = 16 x 1024)
    {
      const int idx = blk * 256 + tid;
      const int b = idx >> 10, j = idx & 1023;
      const float* g = Aa.g1 + b * 4096 + j;
      float cv = sigm(g[1024]) * Aa.c1[idx] + sigm(g[0]) * tanhf(g[2048]);
      float hv = sigm(g[3072]) * tanhf(cv);
      Aa.c1[idx] = cv;
      Aa.h1b[idx] = f2bf(hv);
    }
    gsync(Aa.bar);

    // ---- Phase C1: LSTM2 gates = bias2 + h1*Wih2^T + h2*Whh2^T  (WMMA)
    if (gw < 256) {
      const int n = gw * 16 + li;
      const float bs = Aa.bias2[n];
      v8f acc;
#pragma unroll
      for (int r = 0; r < 8; ++r) acc[r] = bs;
      const u16* ar = Aa.h1b + li * 1024;
      const u16* br = Aa.Wih2 + (size_t)n * 1024;
      __builtin_prefetch(br, 0, 1);
      for (int kb = 0; kb < 1024; kb += 32)
        acc = wmma_bf16(ld_a(ar, kb, half), ld_b(br, kb, half), acc);
      const u16* ar2 = Aa.h2b + li * 1024;
      const u16* br2 = Aa.Whh2 + (size_t)n * 1024;
      for (int kb = 0; kb < 1024; kb += 32)
        acc = wmma_bf16(ld_a(ar2, kb, half), ld_b(br2, kb, half), acc);
#pragma unroll
      for (int r = 0; r < 8; ++r) Aa.g2[(r + 8 * half) * 4096 + n] = acc[r];
    }
    gsync(Aa.bar);

    // ---- Phase C2: LSTM2 pointwise + h2 history
    {
      const int idx = blk * 256 + tid;
      const int b = idx >> 10, j = idx & 1023;
      const float* g = Aa.g2 + b * 4096 + j;
      float cv = sigm(g[1024]) * Aa.c2[idx] + sigm(g[0]) * tanhf(g[2048]);
      float hv = sigm(g[3072]) * tanhf(cv);
      Aa.c2[idx] = cv;
      u16 hb = f2bf(hv);
      Aa.h2b[idx] = hb;
      Aa.H2[(size_t)(t * 16 + b) * 1024 + j] = hb;
    }
    gsync(Aa.bar);
  }
}

// ---------------------------------------------------------------------------
// Host launcher
// ---------------------------------------------------------------------------
extern "C" void kernel_launch(void* const* d_in, const int* in_sizes, int n_in,
                              void* d_out, int out_size, void* d_ws, size_t ws_size,
                              hipStream_t stream) {
  (void)in_sizes; (void)n_in; (void)out_size; (void)ws_size;
  const float* enc    = (const float*)d_in[0];
  const float* spkemb = (const float*)d_in[1];
  const float* mels   = (const float*)d_in[2];
  const float* Wpre1  = (const float*)d_in[3];
  const float* bpre1  = (const float*)d_in[4];
  const float* Wpre2  = (const float*)d_in[5];
  const float* bpre2  = (const float*)d_in[6];
  const float* Wsp    = (const float*)d_in[7];
  const float* bsp    = (const float*)d_in[8];
  const float* Wq     = (const float*)d_in[9];
  const float* Wm     = (const float*)d_in[10];
  const float* Wconv  = (const float*)d_in[11];
  const float* Wloc   = (const float*)d_in[12];
  const float* vatt   = (const float*)d_in[13];
  const float* Wih1   = (const float*)d_in[14];
  const float* Whh1   = (const float*)d_in[15];
  const float* bih1   = (const float*)d_in[16];
  const float* bhh1   = (const float*)d_in[17];
  const float* Wih2   = (const float*)d_in[18];
  const float* Whh2   = (const float*)d_in[19];
  const float* bih2   = (const float*)d_in[20];
  const float* bhh2   = (const float*)d_in[21];
  const float* Wmel   = (const float*)d_in[22];
  const float* bmel   = (const float*)d_in[23];
  const float* Wstop  = (const float*)d_in[24];
  const float* bstop  = (const float*)d_in[25];

  char* base = (char*)d_ws;
  size_t off = 0;
  auto alloc = [&](size_t bytes) -> char* {
    off = (off + 255) & ~(size_t)255;
    char* p = base + off;
    off += bytes;
    return p;
  };
  // bf16 weights
  u16* Wpre1p = (u16*)alloc((size_t)1024 * 96 * 2);
  u16* Wpre2b = (u16*)alloc((size_t)1024 * 1024 * 2);
  u16* Wqb    = (u16*)alloc((size_t)128 * 1024 * 2);
  u16* Wih1pb = (u16*)alloc((size_t)4096 * 1024 * 2);
  u16* Wih1cb = (u16*)alloc((size_t)4096 * 256 * 2);
  u16* Whh1b  = (u16*)alloc((size_t)4096 * 1024 * 2);
  u16* Wih2b  = (u16*)alloc((size_t)4096 * 1024 * 2);
  u16* Whh2b  = (u16*)alloc((size_t)4096 * 1024 * 2);
  u16* Wmelb  = (u16*)alloc((size_t)80 * 1024 * 2);
  u16* Wmb    = (u16*)alloc((size_t)128 * 256 * 2);
  u16* encb   = (u16*)alloc((size_t)4096 * 256 * 2);
  // parallel precompute activations
  u16*   prevmelb = (u16*)alloc((size_t)4096 * 96 * 2);
  u16*   p1b      = (u16*)alloc((size_t)4096 * 1024 * 2);
  u16*   p2b      = (u16*)alloc((size_t)4096 * 1024 * 2);
  u16*   qb       = (u16*)alloc((size_t)4096 * 1024 * 2);
  float* qWb      = (float*)alloc((size_t)4096 * 128 * 4);
  u16*   G1pb     = (u16*)alloc((size_t)4096 * 4096 * 2);
  float* procmem  = (float*)alloc((size_t)4096 * 128 * 4);
  float* spk      = (float*)alloc((size_t)16 * 1024 * 4);
  float* bias1    = (float*)alloc((size_t)4096 * 4);
  float* bias2    = (float*)alloc((size_t)4096 * 4);
  u16*   H2b      = (u16*)alloc((size_t)4096 * 1024 * 2);
  // sequential state (zero-initialized span)
  char* stateBeg = (char*)alloc((size_t)16 * 1024 * 4);  // c1
  float* c1    = (float*)stateBeg;
  float* c2    = (float*)alloc((size_t)16 * 1024 * 4);
  u16*   h1b   = (u16*)alloc((size_t)16 * 1024 * 2);
  u16*   h2b   = (u16*)alloc((size_t)16 * 1024 * 2);
  u16*   ctxb  = (u16*)alloc((size_t)16 * 256 * 2);
  float* prevw = (float*)alloc((size_t)16 * 256 * 4);
  float* cumw  = (float*)alloc((size_t)16 * 256 * 4);
  float* g1    = (float*)alloc((size_t)16 * 4096 * 4);
  float* g2    = (float*)alloc((size_t)16 * 4096 * 4);
  int*   bar   = (int*)alloc(2 * 4);
  size_t stateBytes = (size_t)((base + off) - stateBeg);

  auto gsz = [](int n) { int b = (n + 255) / 256; return b > 8192 ? 8192 : b; };

  // ---- Stage 0: conversions / padding / biases / state init ----------------
  cvt_kernel<<<gsz(1024 * 1024), 256, 0, stream>>>(Wpre2, Wpre2b, 1024 * 1024);
  cvt_kernel<<<gsz(128 * 1024), 256, 0, stream>>>(Wq, Wqb, 128 * 1024);
  cvt_kernel<<<gsz(4096 * 1024), 256, 0, stream>>>(Whh1, Whh1b, 4096 * 1024);
  cvt_kernel<<<gsz(4096 * 1024), 256, 0, stream>>>(Wih2, Wih2b, 4096 * 1024);
  cvt_kernel<<<gsz(4096 * 1024), 256, 0, stream>>>(Whh2, Whh2b, 4096 * 1024);
  cvt_kernel<<<gsz(80 * 1024), 256, 0, stream>>>(Wmel, Wmelb, 80 * 1024);
  cvt_kernel<<<gsz(128 * 256), 256, 0, stream>>>(Wm, Wmb, 128 * 256);
  cvt_kernel<<<gsz(4096 * 256), 256, 0, stream>>>(enc, encb, 4096 * 256);
  split_ih1_kernel<<<gsz(4096 * 1280), 256, 0, stream>>>(Wih1, Wih1pb, Wih1cb, 4096 * 1280);
  pad_pre1_kernel<<<gsz(1024 * 96), 256, 0, stream>>>(Wpre1, Wpre1p);
  prevmel_kernel<<<gsz(4096 * 96), 256, 0, stream>>>(mels, prevmelb);
  bias_sum_kernel<<<gsz(4096), 256, 0, stream>>>(bih1, bhh1, bias1, 4096);
  bias_sum_kernel<<<gsz(4096), 256, 0, stream>>>(bih2, bhh2, bias2, 4096);
  spk_kernel<<<64, 256, 0, stream>>>(Wsp, bsp, spkemb, spk);
  zero_kernel<<<gsz((int)(stateBytes / 4)), 256, 0, stream>>>((int*)stateBeg,
                                                              (int)(stateBytes / 4));

  // ---- Stage 1: parallel WMMA GEMMs over all T*B rows ----------------------
  // prenet1: p1 = relu(prevmel @ Wpre1^T + b1)   M=4096 N=1024 K=96
  gemm_bf16_kernel<2><<<(256 * 64 + 7) / 8, 256, 0, stream>>>(
      prevmelb, 96, Wpre1p, 96, bpre1, nullptr, p1b, nullptr, nullptr,
      4096, 1024, 96, 1024);
  // prenet2: p2 = relu(p1 @ Wpre2^T + b2), q = p2 + spk[b]
  gemm_bf16_kernel<3><<<(256 * 64 + 7) / 8, 256, 0, stream>>>(
      p1b, 1024, Wpre2b, 1024, bpre2, nullptr, p2b, qb, spk,
      4096, 1024, 1024, 1024);
  // qW = q @ Wq^T   (N=128)
  gemm_bf16_kernel<0><<<(256 * 8 + 7) / 8, 256, 0, stream>>>(
      qb, 1024, Wqb, 1024, nullptr, qWb, nullptr, nullptr, nullptr,
      4096, 128, 1024, 128);
  // G1p = p2 @ Wih1[:, :1024]^T + (b_ih1 + b_hh1)   (N=4096, the big one)
  gemm_bf16_kernel<1><<<(256 * 256 + 7) / 8, 256, 0, stream>>>(
      p2b, 1024, Wih1pb, 1024, bias1, nullptr, G1pb, nullptr, nullptr,
      4096, 4096, 1024, 4096);
  // proc_mem = enc @ Wm^T   (rows b*256+l, N=128, K=256)
  gemm_bf16_kernel<0><<<(256 * 8 + 7) / 8, 256, 0, stream>>>(
      encb, 256, Wmb, 256, nullptr, procmem, nullptr, nullptr, nullptr,
      4096, 128, 256, 128);

  // ---- Stage 2: persistent sequential decoder ------------------------------
  DecArgs da;
  da.enc = enc; da.Wc = Wconv; da.Wloc = Wloc; da.vatt = vatt;
  da.qW = qWb; da.procmem = procmem; da.bias2 = bias2;
  da.G1p = G1pb; da.Wih1c = Wih1cb; da.Whh1 = Whh1b;
  da.Wih2 = Wih2b; da.Whh2 = Whh2b;
  da.c1 = c1; da.c2 = c2; da.prevw = prevw; da.cumw = cumw;
  da.g1 = g1; da.g2 = g2;
  da.h1b = h1b; da.h2b = h2b; da.ctxb = ctxb; da.H2 = H2b; da.bar = bar;
  decoder_kernel<<<64, 256, 0, stream>>>(da);

  // ---- Stage 3: output projections -----------------------------------------
  // mel = H2 @ Wmel^T + b_mel  -> d_out[b][t][80]
  gemm_bf16_kernel<4><<<(256 * 5 + 7) / 8, 256, 0, stream>>>(
      H2b, 1024, Wmelb, 1024, bmel, (float*)d_out, nullptr, nullptr, nullptr,
      4096, 80, 1024, 80);
  // stop = sigmoid(H2 . w_stop + b) -> d_out[B*T*80 + b*T + t]
  stop_kernel<<<16, 256, 0, stream>>>(H2b, Wstop, bstop, (float*)d_out);
}
